// MGL4MEP_SRE_17325898072414
// MI455X (gfx1250) — compile-verified, matrix operand order checked by reading, not run on hardware
//
#include <hip/hip_runtime.h>
#include <hip/hip_bf16.h>

typedef __attribute__((ext_vector_type(16))) _Float16 v16h;
typedef __attribute__((ext_vector_type(8)))  _Float16 v8h;
typedef __attribute__((ext_vector_type(8)))  float    v8f;

#define NN 512
#define BB 16
#define TT 12
#define HH 64
#define CIN 66     // DIN + H
#define CPAD 80    // CIN padded to 5*16
#define KPAD 160   // 2*CIN padded to 5*32
#define LH 130

// ---------- WMMA fragment loader ----------
// Row-major matrix (A, or B supplied as B^T row-major N x K), contiguous K.
// Per-lane layout per CDNA5 ISA 7.12.2: lane l holds row l&15; K chunks
// {k0 + 8*(l>=16) + 0..7, k0 + 16 + 8*(l>=16) + 0..7}.
__device__ inline v16h load_frag_rm(const _Float16* __restrict__ base, int ld,
                                    int row0, int k0, int lane) {
  int half = (lane >> 4) & 1;
  int r    = lane & 15;
  const _Float16* p = base + (size_t)(row0 + r) * ld + k0 + half * 8;
  v8h lo = *(const v8h*)(p);
  v8h hi = *(const v8h*)(p + 16);
  v16h out;
#pragma unroll
  for (int j = 0; j < 8; ++j) { out[j] = lo[j]; out[8 + j] = hi[j]; }
  return out;
}

__device__ inline float sigmoidf_(float x) { return 1.f / (1.f + __expf(-x)); }

// ---------- prep kernels ----------

// S_base = softmax(relu(ne @ ne^T)) per row. One block per row.
__global__ void sbase_kernel(const float* __restrict__ ne, float* __restrict__ Sb) {
  __shared__ float vals[NN];
  __shared__ float red[256];
  int m = blockIdx.x, tid = threadIdx.x;
  float nem[10];
#pragma unroll
  for (int d = 0; d < 10; ++d) nem[d] = ne[m * 10 + d];
  for (int j = tid; j < NN; j += 256) {
    float acc = 0.f;
#pragma unroll
    for (int d = 0; d < 10; ++d) acc += nem[d] * ne[j * 10 + d];
    vals[j] = fmaxf(acc, 0.f);
  }
  __syncthreads();
  red[tid] = fmaxf(vals[tid], vals[tid + 256]);
  __syncthreads();
  for (int s = 128; s > 0; s >>= 1) {
    if (tid < s) red[tid] = fmaxf(red[tid], red[tid + s]);
    __syncthreads();
  }
  float rowmax = red[0];
  __syncthreads();
  float e0 = __expf(vals[tid] - rowmax);
  float e1 = __expf(vals[tid + 256] - rowmax);
  vals[tid] = e0; vals[tid + 256] = e1;
  red[tid] = e0 + e1;
  __syncthreads();
  for (int s = 128; s > 0; s >>= 1) {
    if (tid < s) red[tid] += red[tid + s];
    __syncthreads();
  }
  float inv = 1.f / red[0];
  Sb[(size_t)m * NN + tid]       = vals[tid] * inv;
  Sb[(size_t)m * NN + tid + 256] = vals[tid + 256] * inv;
}

// outT[n][o][kk] (f16, kk padded to KPAD) = sum_d ne[n][d] * W[d][kc][i][o]
__global__ void wproj_kernel(const float* __restrict__ ne, const float* __restrict__ W,
                             _Float16* __restrict__ outT, int O, size_t total) {
  size_t idx = (size_t)blockIdx.x * blockDim.x + threadIdx.x;
  if (idx >= total) return;
  int kk = (int)(idx % KPAD);
  int o  = (int)((idx / KPAD) % O);
  int n  = (int)(idx / ((size_t)KPAD * O));
  float v = 0.f;
  if (kk < 2 * CIN) {
    int kc = (kk >= CIN) ? 1 : 0;
    int i  = kk - kc * CIN;
#pragma unroll
    for (int d = 0; d < 10; ++d)
      v += ne[n * 10 + d] * W[(((size_t)d * 2 + kc) * CIN + i) * O + o];
  }
  outT[idx] = (_Float16)v;
}

__global__ void bproj_kernel(const float* __restrict__ ne, const float* __restrict__ bp,
                             float* __restrict__ out, int O) {
  int idx = blockIdx.x * blockDim.x + threadIdx.x;
  int o = idx % O, n = idx / O;
  float v = 0.f;
#pragma unroll
  for (int d = 0; d < 10; ++d) v += ne[n * 10 + d] * bp[d * O + o];
  out[idx] = v;
}

// ---------- per-step kernels ----------

// SmH[b][m][k] = f16(S_base[m][k] * mask[b][t][m][k]); mask is the only
// HBM-resident stream (201 MB total) -> prefetch ahead.
__global__ void smask_kernel(const float* __restrict__ Sb, const float* __restrict__ mask,
                             _Float16* __restrict__ SmH, int t) {
  size_t idx = (size_t)blockIdx.x * blockDim.x + threadIdx.x;   // BB*NN*NN
  size_t rem = idx & (size_t)(NN * NN - 1);
  size_t b   = idx >> 18;   // NN*NN = 1<<18
  const float* mp = mask + (((size_t)b * TT + t) << 18) + rem;
  if (rem + 16384 < (size_t)NN * NN)
    __builtin_prefetch(mp + 16384, 0, 0);
  SmH[idx] = (_Float16)(Sb[rem] * mp[0]);
}

// Build the feature-major B^T operands: insT = concat([x_t, h]) (rows 0..79),
// candT rows {0,1} = x_t and rows 66..79 = 0 (rows 2..65 come from gate_kernel).
__global__ void bmat_kernel(const float* __restrict__ entity, const float* __restrict__ hbuf,
                            _Float16* __restrict__ insT, _Float16* __restrict__ candT, int t) {
  int idx = blockIdx.x * blockDim.x + threadIdx.x;   // BB*CPAD*NN
  int n = idx & (NN - 1);
  int c = (idx >> 9) % CPAD;
  int b = idx / (CPAD * NN);
  float v;
  if (c < 2)        v = entity[((((size_t)b * TT + t) * NN) + n) * 2 + c];
  else if (c < CIN) v = hbuf[((size_t)b * NN + n) * HH + (c - 2)];
  else              v = 0.f;
  insT[idx] = (_Float16)v;
  if (c < 2)         candT[idx] = (_Float16)v;
  else if (c >= CIN) candT[idx] = (_Float16)0.f;
}

// Seed the per-node-layout x_g buffers xg[n][b][KPAD] with the identity-support
// part and zero padding; contiguous stores. The kk in [66,146) region is
// overwritten later by the S@x GEMM epilogue (and [2,66) of xgc by gate_kernel).
__global__ void xgseed_kernel(const float* __restrict__ entity, const float* __restrict__ hbuf,
                              _Float16* __restrict__ xgi, _Float16* __restrict__ xgc, int t) {
  size_t idx = (size_t)blockIdx.x * blockDim.x + threadIdx.x;   // NN*BB*KPAD
  int kk = (int)(idx % KPAD);
  int b  = (int)((idx / KPAD) % BB);
  int n  = (int)(idx / (KPAD * BB));
  float v;
  if (kk < 2)        v = entity[((((size_t)b * TT + t) * NN) + n) * 2 + kk];
  else if (kk < CIN) v = hbuf[((size_t)b * NN + n) * HH + (kk - 2)];
  else               v = 0.f;
  xgi[idx] = (_Float16)v;
  xgc[idx] = (kk < 2) ? (_Float16)v : (_Float16)0.f;
}

// Sx = SmH[b] @ X[b]; per-wave 16x16 tile, K-loop over 512. Epilogue scatters
// into the per-node x_g layout at kk = 66 + feat.
// grid (16 batch, 32 mtiles), blockDim 160 (5 waves = 5 feature tiles).
__global__ void gemm_smask_x(const _Float16* __restrict__ SmH,
                             const _Float16* __restrict__ XT,
                             _Float16* __restrict__ XG) {
  int b    = blockIdx.x;
  int m0   = blockIdx.y * 16;
  int n0   = (threadIdx.x >> 5) * 16;
  int lane = threadIdx.x & 31;
  const _Float16* A  = SmH + (size_t)b * NN * NN;
  const _Float16* Bt = XT  + (size_t)b * CPAD * NN;
  v8f acc = {};
  for (int kt = 0; kt < NN / 32; ++kt) {
    v16h a  = load_frag_rm(A,  NN, m0, kt * 32, lane);
    v16h bm = load_frag_rm(Bt, NN, n0, kt * 32, lane);
    acc = __builtin_amdgcn_wmma_f32_16x16x32_f16(false, a, false, bm, (short)0, acc,
                                                 false, false);
  }
  int feat = n0 + (lane & 15);
  int mb   = (lane >> 4) * 8;
#pragma unroll
  for (int r = 0; r < 8; ++r) {
    int node = m0 + mb + r;
    XG[((size_t)node * BB + b) * KPAD + CIN + feat] = (_Float16)acc[r];
  }
}

// z_r GEMM + sigmoid. grid 512 nodes, 8 waves = 8 output tiles (O=128).
// A-fragments now come from the contiguous xg[n][b][KPAD] layout (b128 loads).
// Writes r into rbuf, and z*h into candT rows 2..65 and xgc kk 2..65.
__global__ void gate_kernel(const _Float16* __restrict__ xgi,
                            const _Float16* __restrict__ WgT, const float* __restrict__ bg,
                            const float* __restrict__ hbuf, float* __restrict__ rbuf,
                            _Float16* __restrict__ candT, _Float16* __restrict__ xgc) {
  int n    = blockIdx.x;
  int ot   = threadIdx.x >> 5;
  int lane = threadIdx.x & 31;
  const _Float16* Arow = xgi + (size_t)n * BB * KPAD;
  const _Float16* Brow = WgT + ((size_t)n * 128 + ot * 16) * KPAD;
  v8f acc = {};
#pragma unroll
  for (int kt = 0; kt < KPAD / 32; ++kt) {
    v16h a  = load_frag_rm(Arow, KPAD, 0, kt * 32, lane);
    v16h bm = load_frag_rm(Brow, KPAD, 0, kt * 32, lane);
    acc = __builtin_amdgcn_wmma_f32_16x16x32_f16(false, a, false, bm, (short)0, acc,
                                                 false, false);
  }
  int o  = ot * 16 + (lane & 15);
  int mb = (lane >> 4) * 8;
#pragma unroll
  for (int r = 0; r < 8; ++r) {
    int b = mb + r;
    float s = sigmoidf_(acc[r] + bg[n * 128 + o]);
    if (o < HH) {
      float zh = s * hbuf[((size_t)b * NN + n) * HH + o];
      candT[((size_t)b * CPAD + 2 + o) * NN + n]  = (_Float16)zh;
      xgc[((size_t)n * BB + b) * KPAD + 2 + o]    = (_Float16)zh;
    } else {
      rbuf[((size_t)b * NN + n) * HH + (o - HH)] = s;
    }
  }
}

// Candidate GEMM + tanh + GRU state update. grid 512 nodes, 4 waves (O=64).
__global__ void update_kernel(const _Float16* __restrict__ xgc,
                              const _Float16* __restrict__ WuT, const float* __restrict__ bu,
                              const float* __restrict__ rbuf, float* __restrict__ hbuf) {
  int n    = blockIdx.x;
  int ot   = threadIdx.x >> 5;
  int lane = threadIdx.x & 31;
  const _Float16* Arow = xgc + (size_t)n * BB * KPAD;
  const _Float16* Brow = WuT + ((size_t)n * 64 + ot * 16) * KPAD;
  v8f acc = {};
#pragma unroll
  for (int kt = 0; kt < KPAD / 32; ++kt) {
    v16h a  = load_frag_rm(Arow, KPAD, 0, kt * 32, lane);
    v16h bm = load_frag_rm(Brow, KPAD, 0, kt * 32, lane);
    acc = __builtin_amdgcn_wmma_f32_16x16x32_f16(false, a, false, bm, (short)0, acc,
                                                 false, false);
  }
  int o  = ot * 16 + (lane & 15);
  int mb = (lane >> 4) * 8;
#pragma unroll
  for (int r = 0; r < 8; ++r) {
    int b = mb + r;
    size_t hi = ((size_t)b * NN + n) * HH + o;
    float hc = tanhf(acc[r] + bu[n * 64 + o]);
    float rv = rbuf[hi];
    hbuf[hi] = rv * hbuf[hi] + (1.f - rv) * hc;
  }
}

// ---------- tail kernels ----------

__global__ void gmax_kernel(const float* __restrict__ hbuf, float* __restrict__ gout) {
  int b = blockIdx.x, o = threadIdx.x;
  float mx = hbuf[((size_t)b * NN) * HH + o];
  for (int n = 1; n < NN; ++n) mx = fmaxf(mx, hbuf[((size_t)b * NN + n) * HH + o]);
  gout[b * HH + o] = mx;
}

// One block per batch element; full 32-step scan of one LSTM layer.
__global__ void lstm_kernel(const float* __restrict__ xin, int in_dim, int xmode,
                            const float* __restrict__ Wih, const float* __restrict__ Whh,
                            const float* __restrict__ bih, const float* __restrict__ bhh,
                            float* __restrict__ seq_out, float* __restrict__ last_out) {
  int b = blockIdx.x, j = threadIdx.x;
  __shared__ float h[LH], c[LH], xv[LH];
  if (j < LH) { h[j] = 0.f; c[j] = 0.f; }
  __syncthreads();
  for (int t = 0; t < 32; ++t) {
    if (j < in_dim)
      xv[j] = (xmode == 0) ? xin[((size_t)b * 32 + t) * 64 + j]
                           : xin[((size_t)t * BB + b) * LH + j];
    __syncthreads();
    float hn = 0.f, cn = 0.f;
    if (j < LH) {
      float gi = bih[j] + bhh[j];
      float gf = bih[LH + j] + bhh[LH + j];
      float gg = bih[2 * LH + j] + bhh[2 * LH + j];
      float go = bih[3 * LH + j] + bhh[3 * LH + j];
      for (int f = 0; f < in_dim; ++f) {
        float x = xv[f];
        gi += x * Wih[(size_t)j * in_dim + f];
        gf += x * Wih[(size_t)(LH + j) * in_dim + f];
        gg += x * Wih[(size_t)(2 * LH + j) * in_dim + f];
        go += x * Wih[(size_t)(3 * LH + j) * in_dim + f];
      }
      for (int f = 0; f < LH; ++f) {
        float hh = h[f];
        gi += hh * Whh[(size_t)j * LH + f];
        gf += hh * Whh[(size_t)(LH + j) * LH + f];
        gg += hh * Whh[(size_t)(2 * LH + j) * LH + f];
        go += hh * Whh[(size_t)(3 * LH + j) * LH + f];
      }
      float iv = sigmoidf_(gi), fv = sigmoidf_(gf);
      float gv = tanhf(gg),     ov = sigmoidf_(go);
      cn = fv * c[j] + iv * gv;
      hn = ov * tanhf(cn);
    }
    __syncthreads();
    if (j < LH) {
      h[j] = hn; c[j] = cn;
      if (seq_out)  seq_out[((size_t)t * BB + b) * LH + j] = hn;
      if (last_out && t == 31) last_out[(size_t)b * LH + j] = hn;
    }
    __syncthreads();
  }
}

__global__ void fc_kernel(const float* __restrict__ h2l, const float* __restrict__ gout,
                          const float* __restrict__ fcW, const float* __restrict__ fcb,
                          float* __restrict__ out) {
  int idx = threadIdx.x;
  if (idx >= BB * 24) return;
  int b = idx / 24, j = idx % 24;
  float acc = fcb[j];
  for (int i = 0; i < LH; ++i) acc += h2l[b * LH + i] * fcW[j * (LH + HH) + i];
  for (int i = 0; i < HH; ++i) acc += gout[b * HH + i] * fcW[j * (LH + HH) + LH + i];
  out[idx] = sigmoidf_(acc);
}

// ---------- launcher ----------

extern "C" void kernel_launch(void* const* d_in, const int* in_sizes, int n_in,
                              void* d_out, int out_size, void* d_ws, size_t ws_size,
                              hipStream_t stream) {
  (void)in_sizes; (void)n_in; (void)out_size; (void)ws_size;
  const float* entity = (const float*)d_in[0];
  const float* mask   = (const float*)d_in[1];
  const float* sr     = (const float*)d_in[2];
  const float* ne     = (const float*)d_in[3];
  const float* gW     = (const float*)d_in[4];
  const float* gb     = (const float*)d_in[5];
  const float* uW     = (const float*)d_in[6];
  const float* ub     = (const float*)d_in[7];
  const float* Wih0   = (const float*)d_in[8];
  const float* Whh0   = (const float*)d_in[9];
  const float* bih0   = (const float*)d_in[10];
  const float* bhh0   = (const float*)d_in[11];
  const float* Wih1   = (const float*)d_in[12];
  const float* Whh1   = (const float*)d_in[13];
  const float* bih1   = (const float*)d_in[14];
  const float* bhh1   = (const float*)d_in[15];
  const float* fcW    = (const float*)d_in[16];
  const float* fcb    = (const float*)d_in[17];
  float* out = (float*)d_out;

  char* ws = (char*)d_ws;
  size_t off = 0;
  auto alloc = [&](size_t bytes) -> char* {
    char* p = ws + off;
    off += (bytes + 255) & ~(size_t)255;
    return p;
  };
  float*    Sb    = (float*)   alloc((size_t)NN * NN * 4);
  _Float16* SmH   = (_Float16*)alloc((size_t)BB * NN * NN * 2);
  _Float16* insT  = (_Float16*)alloc((size_t)BB * CPAD * NN * 2);
  _Float16* candT = (_Float16*)alloc((size_t)BB * CPAD * NN * 2);
  _Float16* xgi   = (_Float16*)alloc((size_t)NN * BB * KPAD * 2);
  _Float16* xgc   = (_Float16*)alloc((size_t)NN * BB * KPAD * 2);
  _Float16* WgT   = (_Float16*)alloc((size_t)NN * 128 * KPAD * 2);
  _Float16* WuT   = (_Float16*)alloc((size_t)NN * 64 * KPAD * 2);
  float* bg   = (float*)alloc((size_t)NN * 128 * 4);
  float* bu   = (float*)alloc((size_t)NN * 64 * 4);
  float* hbuf = (float*)alloc((size_t)BB * NN * HH * 4);
  float* rbuf = (float*)alloc((size_t)BB * NN * HH * 4);
  float* gout = (float*)alloc((size_t)BB * HH * 4);
  float* h1s  = (float*)alloc((size_t)32 * BB * LH * 4);
  float* h2l  = (float*)alloc((size_t)BB * LH * 4);

  hipMemsetAsync(hbuf, 0, (size_t)BB * NN * HH * 4, stream);

  sbase_kernel<<<NN, 256, 0, stream>>>(ne, Sb);
  {
    size_t tot = (size_t)NN * 128 * KPAD;
    wproj_kernel<<<(unsigned)((tot + 255) / 256), 256, 0, stream>>>(ne, gW, WgT, 128, tot);
  }
  {
    size_t tot = (size_t)NN * 64 * KPAD;
    wproj_kernel<<<(unsigned)((tot + 255) / 256), 256, 0, stream>>>(ne, uW, WuT, 64, tot);
  }
  bproj_kernel<<<(NN * 128) / 256, 256, 0, stream>>>(ne, gb, bg, 128);
  bproj_kernel<<<(NN * 64) / 256, 256, 0, stream>>>(ne, ub, bu, 64);

  for (int t = 0; t < TT; ++t) {
    smask_kernel<<<(BB * NN * NN) / 256, 256, 0, stream>>>(Sb, mask, SmH, t);
    bmat_kernel<<<(BB * CPAD * NN) / 256, 256, 0, stream>>>(entity, hbuf, insT, candT, t);
    xgseed_kernel<<<(NN * BB * KPAD) / 256, 256, 0, stream>>>(entity, hbuf, xgi, xgc, t);
    gemm_smask_x<<<dim3(BB, NN / 16), 160, 0, stream>>>(SmH, insT, xgi);
    gate_kernel<<<NN, 256, 0, stream>>>(xgi, WgT, bg, hbuf, rbuf, candT, xgc);
    gemm_smask_x<<<dim3(BB, NN / 16), 160, 0, stream>>>(SmH, candT, xgc);
    update_kernel<<<NN, 128, 0, stream>>>(xgc, WuT, bu, rbuf, hbuf);
  }

  gmax_kernel<<<BB, HH, 0, stream>>>(hbuf, gout);
  lstm_kernel<<<BB, 160, 0, stream>>>(sr, 64, 0, Wih0, Whh0, bih0, bhh0, h1s, nullptr);
  lstm_kernel<<<BB, 160, 0, stream>>>(h1s, LH, 1, Wih1, Whh1, bih1, bhh1, nullptr, h2l);
  fc_kernel<<<1, 384, 0, stream>>>(h2l, gout, fcW, fcb, out);
}
